// RaiseZero_64295660421818
// MI455X (gfx1250) — compile-verified, matrix operand order checked by reading, not run on hardware
//
#include <hip/hip_runtime.h>

// ---------------------------------------------------------------------------
// Types for CDNA5 WMMA (wave32): v_wmma_f32_16x16x32_bf16
// ---------------------------------------------------------------------------
typedef __attribute__((ext_vector_type(16))) __bf16 v16bf;
typedef __attribute__((ext_vector_type(8)))  __bf16 v8bf;
typedef __attribute__((ext_vector_type(8)))  float  v8f;

#define BN_EPS 1e-5f

static __device__ __forceinline__ __bf16 f2bf(float f) {
    unsigned u = __builtin_bit_cast(unsigned, f);
    unsigned r = u + 0x7FFFu + ((u >> 16) & 1u);   // round-to-nearest-even
    unsigned short h = (unsigned short)(r >> 16);
    return __builtin_bit_cast(__bf16, h);
}

static __device__ __forceinline__ float relu(float x) { return fmaxf(x, 0.0f); }

static __device__ __forceinline__ v8f wmma_bf16(v16bf a, v16bf b, v8f c) {
    // D = A(16x32) x B(32x16) + C, f32 accumulate
    return __builtin_amdgcn_wmma_f32_16x16x32_bf16(
        /*neg_a=*/false, a, /*neg_b=*/false, b,
        /*c_mod=*/(short)0, c, /*reuse_a=*/false, /*reuse_b=*/false);
}

// Fragment loader for 16-bit A/B operands (ISA 7.12.2 layout):
// lane L: row = L&15; holds K offsets [8g..8g+7] and [16+8g..16+8g+7], g=L>>4.
// `base` already points at (row0, k0) of a row-major bf16 tile with leading dim `ld`.
static __device__ __forceinline__ v16bf load_frag(const __bf16* base, int ld, int lane) {
    const int m  = lane & 15;
    const int gk = (lane >> 4) << 3;
    const __bf16* p = base + m * ld + gk;
    union { v16bf f; v8bf h[2]; } u;
    u.h[0] = *(const v8bf*)(p);
    u.h[1] = *(const v8bf*)(p + 16);
    return u.f;
}

static __device__ __forceinline__ void wave_lds_fence() {
    asm volatile("s_wait_dscnt 0" ::: "memory");
}

// CDNA5 async copy global->LDS (VGLOBAL encoding, ASYNCcnt-tracked).
static __device__ __forceinline__ void async_ld_b128(unsigned lds_addr, const void* gptr) {
    asm volatile("global_load_async_to_lds_b128 %0, %1, off"
                 :: "v"(lds_addr), "v"(gptr) : "memory");
}
static __device__ __forceinline__ void wait_async0() {
    asm volatile("s_wait_asynccnt 0" ::: "memory");
}

// ---------------------------------------------------------------------------
// Prep kernels: fold BN into weights (bf16, transposed [N][K]) and biases.
// ---------------------------------------------------------------------------
__global__ void zero_f(float* __restrict__ p, long n) {
    long i = (long)blockIdx.x * 256 + threadIdx.x;
    if (i < n) p[i] = 0.0f;
}

// Wt[n*K + k] = bf16( W[k*N + n] * gam[n] * rsqrt(var[n]+eps) )   (gam==null -> s=1)
__global__ void fold_w(const float* __restrict__ W, const float* __restrict__ gam,
                       const float* __restrict__ var, __bf16* __restrict__ Wt,
                       int K, int N) {
    int i = blockIdx.x * 256 + threadIdx.x;
    if (i >= K * N) return;
    int n = i / K, k = i - n * K;
    float s = 1.0f;
    if (gam) s = gam[n] * rsqrtf(var[n] + BN_EPS);
    Wt[i] = f2bf(W[(long)k * N + n] * s);
}

__global__ void fold_b(const float* __restrict__ g, const float* __restrict__ b,
                       const float* __restrict__ m, const float* __restrict__ v,
                       float* __restrict__ c, int N) {
    int i = blockIdx.x * 256 + threadIdx.x;
    if (i < N) c[i] = b[i] - m[i] * g[i] * rsqrtf(v[i] + BN_EPS);
}

__global__ void copy_b(const float* __restrict__ b, float* __restrict__ c, int N) {
    int i = blockIdx.x * 256 + threadIdx.x;
    if (i < N) c[i] = b[i];
}

// ---------------------------------------------------------------------------
// Edge kernel: x = [face_rep[idx], edge_rep] (E x 256)
//   msg = relu(x @ W1' + c1) @ W2' + c2, relu, atomic scatter-add into agg.
// Block = 128 threads (4 waves), 16 edges per wave. Weight slabs staged once
// per block per K-chunk via async global->LDS copies (ASYNCcnt).
// ---------------------------------------------------------------------------
__global__ __launch_bounds__(128) void edge_kernel(
    const float* __restrict__ face_rep, const float* __restrict__ edge_rep,
    const int* __restrict__ idx,
    const __bf16* __restrict__ W1t, const float* __restrict__ c1,   // [512][256], [512]
    const __bf16* __restrict__ W2t, const float* __restrict__ c2,   // [256][512], [256]
    float* __restrict__ agg, long E)
{
    __shared__ __bf16 sh_x[4][16 * 256];   // 32KB: per-wave x staging, later transpose buf
    __shared__ __bf16 sh_w1[32 * 256];     // 16KB: W1' rows [32*nt2, +32) x K=256 (contig)
    __shared__ __bf16 sh_w2[256 * 32];     // 16KB: W2' all 256 rows x 32-col slab, ld=32
    const int lane = threadIdx.x & 31;
    const int wv   = threadIdx.x >> 5;
    const int g    = lane >> 4;
    const int nlo  = lane & 15;
    const long e0  = (long)blockIdx.x * 64 + (long)wv * 16;

    __bf16* shx = sh_x[wv];

    // ---- stage x tile (16 rows x 256 cols) as bf16: lanes 0-15 face half, 16-31 edge half
    for (int r = 0; r < 16; ++r) {
        long e = e0 + r;
        float4 p0 = make_float4(0.f, 0.f, 0.f, 0.f);
        float4 p1 = p0;
        if (e < E) {
            const float* src;
            if (lane < 16) { long f = idx[e]; src = face_rep + f * 128 + lane * 8; }
            else           {                  src = edge_rep + e * 128 + (lane - 16) * 8; }
            p0 = *(const float4*)(src);
            p1 = *(const float4*)(src + 4);
        }
        __bf16* d = shx + r * 256 + lane * 8;
        d[0] = f2bf(p0.x); d[1] = f2bf(p0.y); d[2] = f2bf(p0.z); d[3] = f2bf(p0.w);
        d[4] = f2bf(p1.x); d[5] = f2bf(p1.y); d[6] = f2bf(p1.z); d[7] = f2bf(p1.w);
    }
    wave_lds_fence();

    // ---- A-fragments of x (K = 256 -> 8 chunks of 32)
    v16bf xfrag[8];
#pragma unroll
    for (int kc = 0; kc < 8; ++kc) xfrag[kc] = load_frag(shx + kc * 32, 256, lane);

    __bf16* tb = shx;   // reuse staging space as 16x32 transpose buffer (wave-private)

    // ---- layer-2 accumulators (16 x 16x16 tiles = msg 16x256), bias-initialized
    v8f msgacc[16];
#pragma unroll
    for (int nt = 0; nt < 16; ++nt) {
        float bb = c2[nt * 16 + nlo];
        v8f a = {bb, bb, bb, bb, bb, bb, bb, bb};
        msgacc[nt] = a;
    }

    const unsigned l_w1 = (unsigned)(size_t)(__bf16*)sh_w1;
    const unsigned l_w2 = (unsigned)(size_t)(__bf16*)sh_w2;
    const v8f vzero = {0.f, 0.f, 0.f, 0.f, 0.f, 0.f, 0.f, 0.f};

    // ---- fused layer1 -> transpose -> layer2, weight slabs async-staged per chunk
    for (int nt2 = 0; nt2 < 16; ++nt2) {          // 32-wide chunk of hidden h
        __syncthreads();                           // previous slab fully consumed
        {   // W1' slab: 32 rows x 256 k, contiguous 16KB -> 8 x b128 per thread
            const char* g1p = (const char*)(W1t + (size_t)nt2 * 32 * 256);
#pragma unroll
            for (int j = 0; j < 8; ++j) {
                int off = threadIdx.x * 128 + j * 16;
                async_ld_b128(l_w1 + off, g1p + off);
            }
            // W2' slab: 256 rows x 32 cols (64B per row, row stride 1024B)
            const char* g2p = (const char*)W2t + (size_t)nt2 * 64;
#pragma unroll
            for (int j = 0; j < 8; ++j) {
                int c = threadIdx.x * 8 + j;       // chunk id 0..1023
                int row = c >> 2, part = (c & 3) * 16;
                async_ld_b128(l_w2 + row * 64 + part,
                              g2p + (size_t)row * 1024 + part);
            }
        }
        wait_async0();
        __syncthreads();                           // slab visible to all waves

        // ---- layer 1: 4 independent WMMA chains (2 N-halves x 2 K-halves)
        //      so LDS B-fragment latency is hidden behind 3 other WMMAs.
        {
            const int n0 = nt2 * 32;
            float bb0 = c1[n0 + nlo];
            float bb1 = c1[n0 + 16 + nlo];
            v8f a0 = {bb0, bb0, bb0, bb0, bb0, bb0, bb0, bb0};
            v8f a1 = {bb1, bb1, bb1, bb1, bb1, bb1, bb1, bb1};
            v8f a2 = vzero, a3 = vzero;
#pragma unroll
            for (int kc = 0; kc < 4; ++kc) {
                a0 = wmma_bf16(xfrag[kc],     load_frag(sh_w1 + kc * 32,            256, lane), a0);
                a1 = wmma_bf16(xfrag[kc],     load_frag(sh_w1 + 16 * 256 + kc * 32, 256, lane), a1);
                a2 = wmma_bf16(xfrag[kc + 4], load_frag(sh_w1 + (kc + 4) * 32,      256, lane), a2);
                a3 = wmma_bf16(xfrag[kc + 4], load_frag(sh_w1 + 16 * 256 + (kc + 4) * 32, 256, lane), a3);
            }
#pragma unroll
            for (int v = 0; v < 8; ++v) {          // combine K-halves, ReLU, transpose-store
                tb[(v + 8 * g) * 32 + nlo]      = f2bf(relu(a0[v] + a2[v]));
                tb[(v + 8 * g) * 32 + 16 + nlo] = f2bf(relu(a1[v] + a3[v]));
            }
        }
        wave_lds_fence();
        v16bf hfrag = load_frag(tb, 32, lane);    // A-fragment of h chunk (16x32)

        // ---- layer 2: 16 independent chains, B-fragments in groups of 4
#pragma unroll
        for (int ntg = 0; ntg < 4; ++ntg) {
            v16bf bq[4];
#pragma unroll
            for (int j = 0; j < 4; ++j)
                bq[j] = load_frag(sh_w2 + ((ntg * 4 + j) * 16) * 32, 32, lane);
#pragma unroll
            for (int j = 0; j < 4; ++j)
                msgacc[ntg * 4 + j] = wmma_bf16(hfrag, bq[j], msgacc[ntg * 4 + j]);
        }
    }

    // ---- epilogue: ReLU + scatter-add into agg[face]
#pragma unroll
    for (int v = 0; v < 8; ++v) {
        long e = e0 + v + 8 * g;
        if (e < E) {
            long f = idx[e];
            float* dst = agg + f * 256 + nlo;
#pragma unroll
            for (int nt = 0; nt < 16; ++nt) {
                atomicAdd(dst + nt * 16, relu(msgacc[nt][v]));
            }
        }
    }
}

// ---------------------------------------------------------------------------
// Face kernel: out = relu(relu(agg@W3'+c3) @ W4'+c4) @ W5' + c5
// Block = 64 threads (2 waves), 16 faces per wave. Weights are tiny and
// WGP$-resident; B fragments stream from global (pipelined b128 clauses).
// ---------------------------------------------------------------------------
__global__ __launch_bounds__(64) void face_kernel(
    const float* __restrict__ agg,
    const __bf16* __restrict__ W3t, const float* __restrict__ c3,   // [512][256]
    const __bf16* __restrict__ W4t, const float* __restrict__ c4,   // [512][512]
    const __bf16* __restrict__ W5t, const float* __restrict__ c5,   // [256][512]
    float* __restrict__ out, long F)
{
    __shared__ __bf16 sh_y1[2][16 * 512];   // 32KB: y1 tiles (also agg staging)
    __shared__ __bf16 sh_tb[2][16 * 32];    // 2KB : per-wave chunk transpose
    const int lane = threadIdx.x & 31;
    const int wv   = threadIdx.x >> 5;
    const int g    = lane >> 4;
    const int nlo  = lane & 15;
    const long f0  = (long)blockIdx.x * 32 + (long)wv * 16;

    __bf16* y1 = sh_y1[wv];
    __bf16* tb = sh_tb[wv];
    const v8f vzero = {0.f, 0.f, 0.f, 0.f, 0.f, 0.f, 0.f, 0.f};

    // ---- stage agg tile (16 x 256) as bf16 into front of y1
    for (int r = 0; r < 16; ++r) {
        long f = f0 + r;
        float4 p0 = make_float4(0.f, 0.f, 0.f, 0.f);
        float4 p1 = p0;
        if (f < F) {
            const float* src = agg + f * 256 + lane * 8;
            p0 = *(const float4*)(src);
            p1 = *(const float4*)(src + 4);
        }
        __bf16* d = y1 + r * 256 + lane * 8;
        d[0] = f2bf(p0.x); d[1] = f2bf(p0.y); d[2] = f2bf(p0.z); d[3] = f2bf(p0.w);
        d[4] = f2bf(p1.x); d[5] = f2bf(p1.y); d[6] = f2bf(p1.z); d[7] = f2bf(p1.w);
    }
    wave_lds_fence();

    v16bf xfrag[8];
#pragma unroll
    for (int kc = 0; kc < 8; ++kc) xfrag[kc] = load_frag(y1 + kc * 32, 256, lane);

    // ---- layer 3: y1 = relu(agg @ W3' + c3), 16x512 in LDS;
    //      tiles in pairs, 4 independent chains (2 N x 2 K-halves)
    for (int nt = 0; nt < 32; nt += 2) {
        const int n0 = nt * 16;
        float bb0 = c3[n0 + nlo];
        float bb1 = c3[n0 + 16 + nlo];
        v8f a0 = {bb0, bb0, bb0, bb0, bb0, bb0, bb0, bb0};
        v8f a1 = {bb1, bb1, bb1, bb1, bb1, bb1, bb1, bb1};
        v8f a2 = vzero, a3 = vzero;
#pragma unroll
        for (int kc = 0; kc < 4; ++kc) {
            a0 = wmma_bf16(xfrag[kc],     load_frag(W3t + (long)n0 * 256 + kc * 32,              256, lane), a0);
            a1 = wmma_bf16(xfrag[kc],     load_frag(W3t + (long)(n0 + 16) * 256 + kc * 32,       256, lane), a1);
            a2 = wmma_bf16(xfrag[kc + 4], load_frag(W3t + (long)n0 * 256 + (kc + 4) * 32,        256, lane), a2);
            a3 = wmma_bf16(xfrag[kc + 4], load_frag(W3t + (long)(n0 + 16) * 256 + (kc + 4) * 32, 256, lane), a3);
        }
#pragma unroll
        for (int v = 0; v < 8; ++v) {
            y1[(v + 8 * g) * 512 + n0 + nlo]      = f2bf(relu(a0[v] + a2[v]));
            y1[(v + 8 * g) * 512 + n0 + 16 + nlo] = f2bf(relu(a1[v] + a3[v]));
        }
    }
    wave_lds_fence();

    v16bf y1frag[16];
#pragma unroll
    for (int kc = 0; kc < 16; ++kc) y1frag[kc] = load_frag(y1 + kc * 32, 512, lane);

    v8f outacc[16];
#pragma unroll
    for (int nt = 0; nt < 16; ++nt) {
        float bb = c5[nt * 16 + nlo];
        v8f a = {bb, bb, bb, bb, bb, bb, bb, bb};
        outacc[nt] = a;
    }

    // ---- fused layer4 -> transpose -> layer5
    for (int nt2 = 0; nt2 < 16; ++nt2) {          // 32-wide chunk of y2
#pragma unroll
        for (int half = 0; half < 2; ++half) {
            const int n0 = nt2 * 32 + half * 16;
            float bb = c4[n0 + nlo];
            v8f acc = {bb, bb, bb, bb, bb, bb, bb, bb};
            v8f acc2 = vzero;                      // 2 K-half chains
#pragma unroll
            for (int kc = 0; kc < 8; ++kc) {
                acc  = wmma_bf16(y1frag[kc],     load_frag(W4t + (long)n0 * 512 + kc * 32,       512, lane), acc);
                acc2 = wmma_bf16(y1frag[kc + 8], load_frag(W4t + (long)n0 * 512 + (kc + 8) * 32, 512, lane), acc2);
            }
#pragma unroll
            for (int v = 0; v < 8; ++v) {
                tb[(v + 8 * g) * 32 + half * 16 + nlo] = f2bf(relu(acc[v] + acc2[v]));
            }
        }
        wave_lds_fence();
        v16bf y2frag = load_frag(tb, 32, lane);
#pragma unroll
        for (int ntg = 0; ntg < 4; ++ntg) {        // 16 independent chains, groups of 4
            v16bf bq[4];
#pragma unroll
            for (int j = 0; j < 4; ++j)
                bq[j] = load_frag(W5t + (long)((ntg * 4 + j) * 16) * 512 + nt2 * 32, 512, lane);
#pragma unroll
            for (int j = 0; j < 4; ++j)
                outacc[ntg * 4 + j] = wmma_bf16(y2frag, bq[j], outacc[ntg * 4 + j]);
        }
    }

    // ---- store (final linear: bias already in accumulators, no ReLU)
#pragma unroll
    for (int v = 0; v < 8; ++v) {
        long f = f0 + v + 8 * g;
        if (f < F) {
            float* dst = out + f * 256 + nlo;
#pragma unroll
            for (int nt = 0; nt < 16; ++nt) dst[nt * 16] = outacc[nt][v];
        }
    }
}

// ---------------------------------------------------------------------------
// Launch
// ---------------------------------------------------------------------------
extern "C" void kernel_launch(void* const* d_in, const int* in_sizes, int n_in,
                              void* d_out, int out_size, void* d_ws, size_t ws_size,
                              hipStream_t stream) {
    const float* face_rep = (const float*)d_in[0];
    const float* edge_rep = (const float*)d_in[1];
    const float* W1 = (const float*)d_in[2];
    const float* g1 = (const float*)d_in[3];
    const float* b1 = (const float*)d_in[4];
    const float* m1 = (const float*)d_in[5];
    const float* v1 = (const float*)d_in[6];
    const float* W2 = (const float*)d_in[7];
    const float* g2 = (const float*)d_in[8];
    const float* b2 = (const float*)d_in[9];
    const float* m2 = (const float*)d_in[10];
    const float* v2 = (const float*)d_in[11];
    const float* W3 = (const float*)d_in[12];
    const float* g3 = (const float*)d_in[13];
    const float* b3 = (const float*)d_in[14];
    const float* m3 = (const float*)d_in[15];
    const float* v3 = (const float*)d_in[16];
    const float* W4 = (const float*)d_in[17];
    const float* g4 = (const float*)d_in[18];
    const float* b4 = (const float*)d_in[19];
    const float* m4 = (const float*)d_in[20];
    const float* v4 = (const float*)d_in[21];
    const float* W5 = (const float*)d_in[22];
    const float* bias5 = (const float*)d_in[23];
    const int*   idx = (const int*)d_in[24];

    const long E = in_sizes[1] / 128;   // edges
    const long F = in_sizes[0] / 128;   // faces

    // workspace layout
    char* ws = (char*)d_ws;
    size_t off = 0;
    float* agg = (float*)(ws + off);  off += (size_t)F * 256 * 4;
    off = (off + 255) & ~(size_t)255;
    __bf16* W1t = (__bf16*)(ws + off); off += 512 * 256 * 2;
    __bf16* W2t = (__bf16*)(ws + off); off += 256 * 512 * 2;
    __bf16* W3t = (__bf16*)(ws + off); off += 512 * 256 * 2;
    __bf16* W4t = (__bf16*)(ws + off); off += 512 * 512 * 2;
    __bf16* W5t = (__bf16*)(ws + off); off += 256 * 512 * 2;
    float* c1 = (float*)(ws + off); off += 512 * 4;
    float* c2 = (float*)(ws + off); off += 256 * 4;
    float* c3 = (float*)(ws + off); off += 512 * 4;
    float* c4 = (float*)(ws + off); off += 512 * 4;
    float* c5 = (float*)(ws + off); off += 256 * 4;

    // 1) zero the scatter accumulator
    {
        long n = F * 256;
        zero_f<<<(unsigned)((n + 255) / 256), 256, 0, stream>>>(agg, n);
    }
    // 2) BN-fold + bf16 + transpose weights; fold biases
    fold_w<<<(256 * 512 + 255) / 256, 256, 0, stream>>>(W1, g1, v1, W1t, 256, 512);
    fold_w<<<(512 * 256 + 255) / 256, 256, 0, stream>>>(W2, g2, v2, W2t, 512, 256);
    fold_w<<<(256 * 512 + 255) / 256, 256, 0, stream>>>(W3, g3, v3, W3t, 256, 512);
    fold_w<<<(512 * 512 + 255) / 256, 256, 0, stream>>>(W4, g4, v4, W4t, 512, 512);
    fold_w<<<(512 * 256 + 255) / 256, 256, 0, stream>>>(W5, nullptr, nullptr, W5t, 512, 256);
    fold_b<<<2, 256, 0, stream>>>(g1, b1, m1, v1, c1, 512);
    fold_b<<<1, 256, 0, stream>>>(g2, b2, m2, v2, c2, 256);
    fold_b<<<2, 256, 0, stream>>>(g3, b3, m3, v3, c3, 512);
    fold_b<<<2, 256, 0, stream>>>(g4, b4, m4, v4, c4, 512);
    copy_b<<<1, 256, 0, stream>>>(bias5, c5, 256);

    // 3) edge MLP + scatter
    {
        unsigned blocks = (unsigned)((E + 63) / 64);
        edge_kernel<<<blocks, 128, 0, stream>>>(face_rep, edge_rep, idx,
                                                W1t, c1, W2t, c2, agg, E);
    }
    // 4) face MLP
    {
        unsigned blocks = (unsigned)((F + 31) / 32);
        face_kernel<<<blocks, 64, 0, stream>>>(agg, W3t, c3, W4t, c4, W5t, c5,
                                               (float*)d_out, F);
    }
}
